// SrsRecModel_34565896798471
// MI455X (gfx1250) — matrix operand order repaired and checked by reading, not compiled, not run on hardware
//
#include <hip/hip_runtime.h>
#include <hip/hip_bf16.h>

// ---------------------------------------------------------------------------
// DIN attention, fused for MI455X (gfx1250, wave32, WMMA + TDM).
//   B=4096, L=50, D=64, E=128, CAT=384, U=256
// Strategy:
//   kernel 0: W_hide f32 -> bf16, pre-swizzled into WMMA B-fragment order
//             (one contiguous 16KB slice per K-step -> ideal 1D TDM tile)
//   kernel 1: per block: 2 batches.
//     phase1: gather q/k (coalesced f32), build cat=[q,k,q*k] as bf16 in LDS
//             while wave0's TDM streams W slice 0 into LDS (TENSORcnt)
//     phase2: [128 x 384] @ [384 x 256] via v_wmma_f32_16x16x32_bf16,
//             double-buffered tensor_load_to_lds for W slice ks+1
//     phase3: relu+bias+head fused into epilogue -> scores, mask, out = s@k
// ---------------------------------------------------------------------------

#define B_TOT 4096
#define L_ 50
#define D_ 64
#define E_ 128
#define U_ 256
#define CAT_ 384
#define KSTR 392          // padded lds_cat row stride in halfs (bank-friendly)
#define NKS 12            // 384 / 32 K-steps
#define WSLICE 8192       // halfs per K-step W slice (32x256) = 16KB

typedef __attribute__((ext_vector_type(16))) __bf16 v16bf;
typedef __attribute__((ext_vector_type(8)))  __bf16 v8bf;
typedef __attribute__((ext_vector_type(8)))  float  v8f;
typedef __attribute__((ext_vector_type(4)))  unsigned int u32x4;
typedef __attribute__((ext_vector_type(8)))  int i32x8;
typedef __attribute__((ext_vector_type(4)))  int i32x4;

union ABu { v16bf v; v8bf h[2]; };

// TDM availability / arity selection (see CDNA5_HIP.md: ROCm7.2 = 5 args,
// amdgpu-toolchain (ships the TDM header) = 6 args).
#if defined(__HIP_DEVICE_COMPILE__) && __has_builtin(__builtin_amdgcn_tensor_load_to_lds)
  #define HAVE_TDM 1
#else
  #define HAVE_TDM 0
#endif
#if __has_include(<hip/amd_detail/amd_gfx1250_TDM.h>)
  #define TDM_ARGS6 1
#else
  #define TDM_ARGS6 0
#endif

__device__ __forceinline__ unsigned short f2bf(float f) {
    unsigned int u = __float_as_uint(f);
    unsigned int r = u + 0x7FFFu + ((u >> 16) & 1u);   // round-to-nearest-even
    return (unsigned short)(r >> 16);
}

#if HAVE_TDM
// 1D contiguous DMA: ndw dwords global -> LDS. D# per ISA ch.8:
//  g0: count=1 | lds_addr | global_addr[56:0] | type=2
//  g1: data_size=4B, tensor_dim0=tile_dim0=ndw, tensor_dim1=1, stride0=ndw
__device__ __forceinline__ void tdm_load_1d(unsigned int lds_addr,
                                            const void* gptr, int ndw) {
    unsigned long long ga = (unsigned long long)(uintptr_t)gptr;
    u32x4 g0;
    g0.x = 1u;                                            // count=1, user load
    g0.y = lds_addr;                                      // LDS byte address
    g0.z = (unsigned int)ga;                              // global_addr[31:0]
    g0.w = ((unsigned int)(ga >> 32) & 0x1FFFFFFu) | (2u << 30); // [56:32]|type=2
    i32x8 g1;
    g1[0] = (2 << 16);                    // data_size code 2 = 4 bytes
    g1[1] = (ndw & 0xFFFF) << 16;         // tensor_dim0[15:0]
    g1[2] = (1 << 16) | ((ndw >> 16) & 0xFFFF); // tensor_dim1=1 | tensor_dim0 hi
    g1[3] = (ndw & 0xFFFF) << 16;         // tile_dim0 (16-bit)
    g1[4] = 0;                            // tile_dim1/2 unused
    g1[5] = ndw;                          // tensor_dim0_stride[31:0]
    g1[6] = 0;
    g1[7] = 0;
    i32x4 gz = {0, 0, 0, 0};
#if TDM_ARGS6
    i32x8 gz8 = {0, 0, 0, 0, 0, 0, 0, 0};
    __builtin_amdgcn_tensor_load_to_lds(g0, g1, gz, gz, gz8, 0);
#else
    __builtin_amdgcn_tensor_load_to_lds(g0, g1, gz, gz, 0);
#endif
}
#endif // HAVE_TDM

// ---------------------------------------------------------------------------
// Kernel 0: convert W_hide [384,256] f32 row-major -> bf16 in B-fragment order:
//   ws[ ((ks*16 + nt)*32 + lane)*16 + j ] = bf16( W[k][n] )
//   n = nt*16 + (lane&15), k = ks*32 + (lane>>4)*16 + j
// B-fragment k order used here (contiguous, same as A): lanes 0-15 hold K=0..15, lanes 16-31 hold
// K=16..31, N=lane&15; 16 halfs contiguous per lane. Each K-step slice is a
// contiguous 16KB run -> 1D TDM tile.
// ---------------------------------------------------------------------------
__global__ void w_swizzle_kernel(const float* __restrict__ W,
                                 unsigned short* __restrict__ ws) {
    int idx = blockIdx.x * 256 + threadIdx.x;        // 0 .. 98303
    if (idx >= NKS * 16 * 32 * 16) return;
    int j    = idx & 15;
    int lane = (idx >> 4) & 31;
    int nt   = (idx >> 9) & 15;
    int ks   = idx >> 13;
    int n = nt * 16 + (lane & 15);
    int k = ks * 32 + ((lane >> 4) << 4) + j;
    ws[idx] = f2bf(W[k * U_ + n]);
}

// ---------------------------------------------------------------------------
// Fused DIN kernel. One block = 2 batches, 256 threads = 8 waves.
// ---------------------------------------------------------------------------
__launch_bounds__(256)
__global__ void din_fused_kernel(const int*  __restrict__ qii,
                                 const int*  __restrict__ qic,
                                 const int*  __restrict__ sii,
                                 const int*  __restrict__ sic,
                                 const unsigned char* __restrict__ mask, // jax bool
                                 const float* __restrict__ ei,
                                 const float* __restrict__ ec,
                                 const unsigned short* __restrict__ wsw, // swizzled W bf16
                                 const float* __restrict__ bh,
                                 const float* __restrict__ wo,
                                 const float* __restrict__ bo,
                                 float* __restrict__ out) {
    __shared__ __align__(16) unsigned short lds_cat[128][KSTR]; // bf16 cat rows
    __shared__ float lds_k[2][L_][E_];                          // f32 k copy
    __shared__ float lds_q[2][E_];
    __shared__ float lds_sc[128];
#if HAVE_TDM
    __shared__ __align__(16) unsigned short lds_wb[2][WSLICE];  // W double buffer
#endif

    const int tid  = threadIdx.x;
    const int lane = tid & 31;
    const int wave = tid >> 5;
    const int b0   = blockIdx.x * 2;

#if HAVE_TDM
    const unsigned int wb_addr[2] = {
        (unsigned int)(uintptr_t)&lds_wb[0][0],
        (unsigned int)(uintptr_t)&lds_wb[1][0] };
    // Kick off DMA of W slice 0 (wave 0 only); overlaps the gather phase.
    if (tid < 32)
        tdm_load_1d(wb_addr[0], wsw, WSLICE / 2);
#endif

    // ---------------- phase 1a: q gather, scores init, pad-row zero --------
    {
        int bb = tid >> 7, e = tid & 127;
        int b  = b0 + bb;
        int id = (e < D_) ? qii[b] : qic[b];
        const float* emb = (e < D_) ? ei : ec;
        lds_q[bb][e] = emb[id * D_ + (e & (D_ - 1))];
    }
    if (tid < 128) lds_sc[tid] = 0.0f;
    for (int i = tid; i < 2 * 14 * CAT_; i += 256) {   // zero rows l=50..63
        int bb  = i / (14 * CAT_);
        int rem = i - bb * (14 * CAT_);
        int l   = L_ + rem / CAT_;
        int c   = rem % CAT_;
        lds_cat[bb * 64 + l][c] = 0;
    }
    __syncthreads();

    // ---------------- phase 1b: k gather + cat = [q, k, q*k] ---------------
    for (int i = tid; i < 2 * L_ * E_; i += 256) {
        int e  = i & 127;
        int bl = i >> 7;                 // 0..99
        int bb = (bl >= L_) ? 1 : 0;
        int l  = bl - bb * L_;
        int b  = b0 + bb;
        int id = (e < D_) ? sii[b * L_ + l] : sic[b * L_ + l];
        const float* emb = (e < D_) ? ei : ec;
        float kv = emb[id * D_ + (e & (D_ - 1))];
        float qv = lds_q[bb][e];
        lds_k[bb][l][e] = kv;
        int r = bb * 64 + l;
        lds_cat[r][e]        = f2bf(qv);
        lds_cat[r][E_ + e]   = f2bf(kv);
        lds_cat[r][2*E_ + e] = f2bf(qv * kv);
    }
#if HAVE_TDM
    if (tid < 32) __builtin_amdgcn_s_wait_tensorcnt(0);  // slice 0 landed
#endif
    __syncthreads();

    // ---------------- phase 2: [128x384] @ [384x256] on WMMA ---------------
    // wave w owns N-tiles {2w, 2w+1}, all 8 M-tiles -> 16 accum tiles.
    const int nt0   = wave * 2;
    const int khalf = lane >> 4;        // A frag: 0 -> K{0-15}, 1 -> K{16-31}
    const int mrow  = lane & 15;

    v8f acc[2][8];
    const v8f vzero = {0.f, 0.f, 0.f, 0.f, 0.f, 0.f, 0.f, 0.f};
    #pragma unroll
    for (int j = 0; j < 2; ++j)
        #pragma unroll
        for (int mt = 0; mt < 8; ++mt) acc[j][mt] = vzero;

    for (int ks = 0; ks < NKS; ++ks) {
#if HAVE_TDM
        // prefetch W slice ks+1 into the other LDS buffer via TDM
        if (ks + 1 < NKS && tid < 32)
            tdm_load_1d(wb_addr[(ks + 1) & 1], wsw + (ks + 1) * WSLICE, WSLICE / 2);
        const unsigned short* bbase = &lds_wb[ks & 1][0];
        const v16bf bf0 = *(const v16bf*)(bbase + (((nt0    ) * 32 + lane) << 4));
        const v16bf bf1 = *(const v16bf*)(bbase + (((nt0 + 1) * 32 + lane) << 4));
#else
        if (ks + 1 < NKS)
            __builtin_prefetch(wsw + ((((ks + 1) * 16 + nt0) * 32 + lane) << 4), 0, 1);
        const v16bf bf0 = *(const v16bf*)(wsw + (((ks * 16 + nt0    ) * 32 + lane) << 4));
        const v16bf bf1 = *(const v16bf*)(wsw + (((ks * 16 + nt0 + 1) * 32 + lane) << 4));
#endif
        const int klo = ks * 32 + (khalf << 4);
        #pragma unroll
        for (int mt = 0; mt < 8; ++mt) {
            const unsigned short* ap = &lds_cat[mt * 16 + mrow][klo];
            ABu a;
            a.h[0] = *(const v8bf*)(ap);        // K offsets +0..7   (hi lanes +16..23)
            a.h[1] = *(const v8bf*)(ap + 8);    // K offsets +8..15  (hi lanes +24..31)
            acc[0][mt] = __builtin_amdgcn_wmma_f32_16x16x32_bf16(
                false, a.v, false, bf0, (short)0, acc[0][mt], false, false);
            acc[1][mt] = __builtin_amdgcn_wmma_f32_16x16x32_bf16(
                false, a.v, false, bf1, (short)0, acc[1][mt], false, false);
        }
#if HAVE_TDM
        if (ks + 1 < NKS) {
            if (tid < 32) __builtin_amdgcn_s_wait_tensorcnt(0);
            __syncthreads();    // next slice visible to all waves
        }
#endif
    }

    // ---------------- epilogue: relu(h+b)·W_out, reduce over N -------------
    {
        int n0 = nt0 * 16 + (lane & 15);
        int n1 = n0 + 16;
        float bh0 = bh[n0], wo0 = wo[n0];
        float bh1 = bh[n1], wo1 = wo[n1];
        #pragma unroll
        for (int mt = 0; mt < 8; ++mt) {
            #pragma unroll
            for (int r = 0; r < 8; ++r) {
                float h0 = acc[0][mt][r] + bh0; h0 = h0 > 0.f ? h0 : 0.f;
                float h1 = acc[1][mt][r] + bh1; h1 = h1 > 0.f ? h1 : 0.f;
                float s  = h0 * wo0 + h1 * wo1;
                int   m  = mt * 16 + (khalf << 3) + r;   // C layout: lanes>=16 -> M+8
                atomicAdd(&lds_sc[m], s);                 // ds_add_f32
            }
        }
    }
    __syncthreads();

    // ---------------- phase 3: mask, out[b] = scores @ k -------------------
    if (tid < 128) {
        int bb = tid >> 6, l = tid & 63;
        float s = 0.0f;
        if (l < L_) {
            int b = b0 + bb;
            float mk = mask[b * L_ + l] ? 1.0f : 0.0f;
            s = (lds_sc[tid] + bo[0]) * mk;
        }
        lds_sc[tid] = s;
    }
    __syncthreads();
    {
        int bb = tid >> 7, e = tid & 127;
        float o = 0.0f;
        #pragma unroll 5
        for (int l = 0; l < L_; ++l)
            o += lds_sc[bb * 64 + l] * lds_k[bb][l][e];
        out[(b0 + bb) * E_ + e] = o;
    }
}

// ---------------------------------------------------------------------------
extern "C" void kernel_launch(void* const* d_in, const int* in_sizes, int n_in,
                              void* d_out, int out_size, void* d_ws, size_t ws_size,
                              hipStream_t stream) {
    const int*   qii  = (const int*)d_in[0];
    const int*   qic  = (const int*)d_in[1];
    const int*   sii  = (const int*)d_in[2];
    const int*   sic  = (const int*)d_in[3];
    const unsigned char* mask = (const unsigned char*)d_in[4]; // bool
    const float* ei   = (const float*)d_in[5];
    const float* ec   = (const float*)d_in[6];
    const float* Wh   = (const float*)d_in[7];
    const float* bh   = (const float*)d_in[8];
    const float* wo   = (const float*)d_in[9];
    const float* bo   = (const float*)d_in[10];
    float* out = (float*)d_out;

    unsigned short* wsw = (unsigned short*)d_ws;   // 384*256*2 = 196608 B

    const int B = in_sizes[0];                     // 4096

    // W_hide -> bf16 swizzled B-fragments (K-step slices contiguous)
    w_swizzle_kernel<<<(NKS * 16 * 32 * 16 + 255) / 256, 256, 0, stream>>>(Wh, wsw);

    // fused DIN: 2 batches per block
    din_fused_kernel<<<(B + 1) / 2, 256, 0, stream>>>(
        qii, qic, sii, sic, mask, ei, ec, wsw, bh, wo, bo, out);
}
